// ConvolutionalCapsule_62234076119300
// MI455X (gfx1250) — matrix operand order repaired and checked by reading, not compiled, not run on hardware
//
#include <hip/hip_runtime.h>
#include <hip/hip_bf16.h>
#include <math.h>

#define EPS    1e-9f
#define I_TOT  288
#define O_TOT  32
#define VSTR   514   // i-stride of votes in LDS (bf16 elems); 1028B = 257 dwords -> lane bank stride 1

typedef __attribute__((ext_vector_type(16))) __bf16 v16bf;
typedef __attribute__((ext_vector_type(8)))  float  v8f;

__device__ __forceinline__ float bits2f(unsigned int b) {
  union { unsigned int i; float f; } x; x.i = b; return x.f;
}
__device__ __forceinline__ float bf2f(unsigned short u) {
  return bits2f(((unsigned int)u) << 16);
}
__device__ __forceinline__ unsigned short f2bf(float f) {
  union { float f; unsigned int i; } x; x.f = f;
  unsigned int i = x.i;
  unsigned int r = (i + 0x7FFFu + ((i >> 16) & 1u)) >> 16;  // round-to-nearest-even
  return (unsigned short)r;
}

// ---------------------------------------------------------------------------
// Pack w [288,32,4,4] f32 -> WMMA B-fragment layout, bf16.
// Fragment f = ig*8 + ot covers i in [ig*4, ig*4+4), o in [ot*4, ot*4+4).
// Only K=0..15 are stored (lane half h loads k 8h..8h+7); K=16..31 are zero and
// synthesized at load time.  packed[f*256 + lane*16 + k] =
//     w[ig*4 + k/4][ot*4 + lane/4][k%4][lane%4]
// ---------------------------------------------------------------------------
__global__ void pack_w_kernel(const float* __restrict__ w,
                              unsigned short* __restrict__ packed) {
  int f = blockIdx.x;          // 0..575
  int t = threadIdx.x;         // 0..255
  int lane = t >> 4, k = t & 15;
  int ig = f >> 3, ot = f & 7;
  int i = ig * 4 + (k >> 2);
  int o = ot * 4 + (lane >> 2);
  int q = k & 3, r = lane & 3;
  float val = w[((i * 32 + o) << 4) + q * 4 + r];
  packed[(f << 8) + (lane << 4) + k] = f2bf(val);
}

// ---------------------------------------------------------------------------
// Fused capsule conv + EM routing.  One workgroup (256 thr / 8 waves) per
// output spatial position; votes live in LDS (bf16) for all 3 iterations.
// ---------------------------------------------------------------------------
__global__ void __launch_bounds__(256)
capsule_em_kernel(const float* __restrict__ gpose,
                  const float* __restrict__ gact,
                  const float* __restrict__ beta_v,
                  const float* __restrict__ beta_a,
                  const unsigned short* __restrict__ packedW,
                  float* __restrict__ out) {
  extern __shared__ char smem[];
  unsigned short* v_lds    = (unsigned short*)smem;                 // [288][514] bf16 votes
  unsigned short* rr_lds   = (unsigned short*)(smem + 296064);      // [288][32] bf16
  char*           aux      = smem + 314496;                         // 9216B region
  unsigned short* pose_lds = (unsigned short*)aux;                  // [288][16] bf16 (phase1 only)
  float* mean_lds  = (float*)aux;                                   // [32][16] (aliases pose)
  float* stdv_lds  = (float*)(aux + 2048);                          // [32][16]
  float* rrsum_lds = (float*)(aux + 4096);                          // [32]
  float* logs_lds  = (float*)(aux + 4224);                          // [32]
  float* oact_lds  = (float*)(aux + 4352);                          // [32]
  float* logz_lds  = (float*)(aux + 4480);                          // [32]
  float* i2v_lds   = (float*)(aux + 4608);                          // [32][16]
  float* act_lds   = (float*)(smem + 323712);                       // [288] f32

  int n   = blockIdx.x;                 // 0..575 = ((b*12)+y)*12+x
  int b   = n / 144, y = (n / 12) % 12, x = n % 12;
  int tid = threadIdx.x;

  // ---- Phase 0: stage pose patch (bf16), activations (f32), init rr ----
  for (int e = tid; e < I_TOT * 16; e += 256) {
    int i = e >> 4, m = e & 15;
    int ky = i / 96, kx = (i % 96) >> 5, c = i & 31;
    float v = gpose[((((b * 14) + y + ky) * 14 + x + kx) * 32 + c) * 16 + m];
    pose_lds[e] = f2bf(v);
  }
  for (int i = tid; i < I_TOT; i += 256) {
    int ky = i / 96, kx = (i % 96) >> 5, c = i & 31;
    act_lds[i] = gact[(((b * 14) + y + ky) * 14 + x + kx) * 32 + c];
  }
  for (int e = tid; e < I_TOT * O_TOT; e += 256) rr_lds[e] = 0x3D00;  // bf16(1/32)
  __syncthreads();

  // ---- Phase 1: votes via v_wmma_f32_16x16x32_bf16 ----
  // A (16x32): rows (ii,p) for 4 capsules, block-diagonal in K[0,16); K>=16 zero.
  // B (32x16): K=(ii,q) rows x (oo,r) cols from packed weights; K>=16 zero.
  int wid = tid >> 5, lane = tid & 31;
  int lgrp = lane >> 4, Ml = lane & 15, ii = Ml >> 2, p = Ml & 3;
  bool alane = (lgrp == 0) ? (ii < 2) : (ii >= 2);
  bool hihalf = (ii & 1) != 0;          // which dword pair of the fragment holds our 4 values
  if (lane == 0) __builtin_prefetch(packedW + (wid * 9 * 8) * 256, 0, 3);

  for (int ig = wid * 9; ig < wid * 9 + 9; ++ig) {     // 8 waves x 9 = 72 i-groups
    // one aligned 8B LDS read; fragment built with constant-index dword writes
    uint2 pv = make_uint2(0u, 0u);
    if (alane) pv = *(const uint2*)(pose_lds + ((ig * 4 + ii) << 4) + (p << 2));
    union { v16bf v; unsigned int w[8]; } au;
    au.w[0] = hihalf ? 0u : pv.x;
    au.w[1] = hihalf ? 0u : pv.y;
    au.w[2] = hihalf ? pv.x : 0u;
    au.w[3] = hihalf ? pv.y : 0u;
    au.w[4] = 0u; au.w[5] = 0u; au.w[6] = 0u; au.w[7] = 0u;

    for (int ot = 0; ot < 8; ++ot) {
      union { v16bf v; unsigned int w[8]; } bu;
      {
        uint4 bt = *(const uint4*)(packedW + (((ig * 8 + ot) << 8) + ((lane & 15) << 4) + (lgrp << 3)));
        bu.w[0] = bt.x; bu.w[1] = bt.y; bu.w[2] = bt.z; bu.w[3] = bt.w;
        bu.w[4] = 0u; bu.w[5] = 0u; bu.w[6] = 0u; bu.w[7] = 0u;
      }
      v8f c = {};
      v8f d = __builtin_amdgcn_wmma_f32_16x16x32_bf16(
          false, au.v, false, bu.v, (short)0, c, false, false);
      #pragma unroll
      for (int j = 0; j < 8; ++j) {                    // D: lanes 0-15 M=j, 16-31 M=8+j
        int Mr   = (lgrp << 3) + j;
        int irow = ig * 4 + (Mr >> 2);
        int m    = ((Mr & 3) << 2) + (lane & 3);
        int o    = (ot << 2) + ((lane & 15) >> 2);
        v_lds[irow * VSTR + (o << 4) + m] = f2bf(d[j]);
      }
    }
  }
  __syncthreads();

  // ---- Phase 2: EM routing (3 iterations), votes resident in LDS ----
  float bv = 0.f, ba = 0.f;
  if (tid < 32) { bv = beta_v[tid]; ba = beta_a[tid]; }

  int om0 = tid << 1;                   // adjacent (o,m) pair per thread
  int oo  = tid >> 3;                   // o of that pair

  for (int it = 0; it < 3; ++it) {
    float inv_temp = 1.0f + (float)it;  // 1,2,3

    // step A: single fused pass: Sum(rr'), Sum(rr'*v), Sum(rr'*v^2)
    float s1a = 0.f, s1b = 0.f, s2a = 0.f, s2b = 0.f, wsum = 0.f;
    for (int i = 0; i < I_TOT; ++i) {
      float rrp = bf2f(rr_lds[(i << 5) + oo]) * act_lds[i];
      unsigned int pv = *(const unsigned int*)(v_lds + i * VSTR + om0);
      float v0 = bits2f(pv << 16);
      float v1 = bits2f(pv & 0xFFFF0000u);
      float t0 = rrp * v0, t1 = rrp * v1;
      s1a += t0;               s1b += t1;
      s2a = fmaf(t0, v0, s2a); s2b = fmaf(t1, v1, s2b);
      wsum += rrp;
    }
    if ((tid & 7) == 0) rrsum_lds[oo] = wsum;
    __syncthreads();
    {
      float rs  = rrsum_lds[oo];
      float mu0 = s1a / rs, mu1 = s1b / rs;
      float va0 = fmaxf(s2a / rs - mu0 * mu0, 0.f);
      float va1 = fmaxf(s2b / rs - mu1 * mu1, 0.f);
      float sv0 = sqrtf(va0), sv1 = sqrtf(va1);
      mean_lds[om0]     = mu0;  mean_lds[om0 + 1] = mu1;
      stdv_lds[om0]     = sv0;  stdv_lds[om0 + 1] = sv1;
      i2v_lds[om0]      = 0.5f / (sv0 * sv0);
      i2v_lds[om0 + 1]  = 0.5f / (sv1 * sv1);
    }
    __syncthreads();

    // step B: per-o cost -> activation
    if (tid < 32) {
      float logs = 0.f;
      #pragma unroll
      for (int m = 0; m < 16; ++m) logs += __logf(stdv_lds[(tid << 4) + m] + EPS);
      float cost = rrsum_lds[tid] * (16.0f * bv + logs);
      float a = 1.0f / (1.0f + __expf(-inv_temp * (ba - cost)));
      oact_lds[tid] = a;
      logz_lds[tid] = __logf(a + EPS);
      logs_lds[tid] = logs;
    }
    __syncthreads();

    // step C: rr <- softmax_o( log(act)+o_p ).  One thread per input capsule.
    if (it < 2) {
      for (int rep = 0; rep < 2; ++rep) {
        int i = tid + (rep << 8);
        if (i < I_TOT) {
          float zz[32]; float zmax = -1e30f;
          for (int o = 0; o < O_TOT; ++o) {
            float ssum = 0.f;
            #pragma unroll
            for (int mh = 0; mh < 8; ++mh) {
              int om = (o << 4) + (mh << 1);
              unsigned int pv = *(const unsigned int*)(v_lds + i * VSTR + om);
              float d0 = bits2f(pv << 16)        - mean_lds[om];
              float d1 = bits2f(pv & 0xFFFF0000u) - mean_lds[om + 1];
              ssum = fmaf(d0 * d0, i2v_lds[om], ssum);
              ssum = fmaf(d1 * d1, i2v_lds[om + 1], ssum);
            }
            float z = logz_lds[o] - ssum - logs_lds[o];
            zz[o] = z;
            zmax = fmaxf(zmax, z);
          }
          float se = 0.f;
          #pragma unroll
          for (int o = 0; o < O_TOT; ++o) { float e = __expf(zz[o] - zmax); zz[o] = e; se += e; }
          float inv = 1.0f / se;
          #pragma unroll
          for (int o = 0; o < O_TOT; ++o) rr_lds[(i << 5) + o] = f2bf(zz[o] * inv);
        }
      }
      __syncthreads();
    }
  }

  // ---- Output: pose (mean) then activation, concatenated flat ----
  {
    float2 po;
    po.x = mean_lds[om0];
    po.y = mean_lds[om0 + 1];
    *(float2*)(out + (n << 9) + om0) = po;
  }
  if (tid < 32) out[294912 + (n << 5) + tid] = oact_lds[tid];
}

extern "C" void kernel_launch(void* const* d_in, const int* in_sizes, int n_in,
                              void* d_out, int out_size, void* d_ws, size_t ws_size,
                              hipStream_t stream) {
  (void)in_sizes; (void)n_in; (void)out_size; (void)ws_size;
  const float* gpose = (const float*)d_in[0];
  const float* gact  = (const float*)d_in[1];
  const float* w     = (const float*)d_in[2];
  const float* bv    = (const float*)d_in[3];
  const float* ba    = (const float*)d_in[4];
  // d_in[5]=stride(1), d_in[6]=routings(3): fixed by the reference config.

  unsigned short* packed = (unsigned short*)d_ws;  // 576*256*2 = 294,912 B

  static const int SMEM = 324864;                  // < 320KB WGP LDS
  hipFuncSetAttribute((const void*)capsule_em_kernel,
                      hipFuncAttributeMaxDynamicSharedMemorySize, SMEM);

  pack_w_kernel<<<576, 256, 0, stream>>>(w, packed);
  capsule_em_kernel<<<576, 256, SMEM, stream>>>(gpose, gact, bv, ba, packed,
                                                (float*)d_out);
}
